// GATConv_16604343566548
// MI455X (gfx1250) — matrix operand order repaired and checked by reading, not compiled, not run on hardware
//
#include <hip/hip_runtime.h>

typedef __bf16        v16bf __attribute__((ext_vector_type(16)));
typedef __bf16        v8bf  __attribute__((ext_vector_type(8)));
typedef float         v8f   __attribute__((ext_vector_type(8)));
typedef unsigned int  v4u   __attribute__((ext_vector_type(4)));
typedef int           v4i   __attribute__((ext_vector_type(4)));
typedef int           v8i   __attribute__((ext_vector_type(8)));

#define DIN   128
#define HOUT  512          // H * DOUT
#define NHEAD 4
#define DOUTC 128
#define SLOPE 0.2f

// ---------------------------------------------------------------------------
// init: zero output accumulator, set segment-max to -inf, denom to 0
// ---------------------------------------------------------------------------
__global__ __launch_bounds__(256) void gat_init(float* __restrict__ out,
                                                float* __restrict__ mx,
                                                float* __restrict__ denom,
                                                int nOut, int nNH) {
  int i = blockIdx.x * blockDim.x + threadIdx.x;
  if (i < nOut) out[i] = 0.0f;
  if (i < nNH) {
    mx[i] = __int_as_float(0xff800000);   // -inf
    denom[i] = 0.0f;
  }
}

// ---------------------------------------------------------------------------
// pack x (f32 row-major) -> bf16 row-major
// ---------------------------------------------------------------------------
__global__ __launch_bounds__(256) void gat_pack_x(const float* __restrict__ x,
                                                  __bf16* __restrict__ xb,
                                                  int n) {
  int i = blockIdx.x * blockDim.x + threadIdx.x;
  if (i < n) xb[i] = (__bf16)x[i];
}

// ---------------------------------------------------------------------------
// pack W (f32 [128,512] row-major) -> bf16 in WMMA-B fragment layout:
//   Wp[((nt*4 + kblk)*32 + lane)*16 + j] = W[kblk*32 + (lane<16?0:16) + j,
//                                           nt*16 + (lane&15)]
// so a wave reads its 32x16 bf16 B-fragment as one contiguous 1KB block.
// ---------------------------------------------------------------------------
__global__ __launch_bounds__(256) void gat_pack_w(const float* __restrict__ W,
                                                  __bf16* __restrict__ Wp) {
  int o = blockIdx.x * blockDim.x + threadIdx.x;
  if (o >= DIN * HOUT) return;
  int j    = o & 15;
  int lane = (o >> 4) & 31;
  int kblk = (o >> 9) & 3;
  int nt   = o >> 11;
  int k    = kblk * 32 + ((lane < 16) ? 0 : 16) + j;
  int col  = nt * 16 + (lane & 15);
  Wp[o] = (__bf16)W[(size_t)k * HOUT + col];
}

// ---------------------------------------------------------------------------
// h = x @ W via v_wmma_f32_16x16x32_bf16.
// Block = 4 waves, all sharing one 64-col super-tile (ns); wave w -> m-tile
// mg*4+w.  The block's 16KB of packed B is TDM-DMA'd into LDS once
// (tensor_load_to_lds + s_wait_tensorcnt), then each wave runs a clean
// K-loop: 2x b128 A loads (pre-converted bf16, row clamped -> no predication),
// 4x LDS B fragments, 4x WMMA.
// ---------------------------------------------------------------------------
__global__ __launch_bounds__(128) void gat_gemm_wmma(const __bf16* __restrict__ xb,
                                                     const __bf16* __restrict__ Wp,
                                                     float* __restrict__ h,
                                                     int Nn) {
  __shared__ __attribute__((aligned(32))) __bf16 Bsh[8192];   // 16 KB

  const int lane = threadIdx.x & 31;
  const int wave = threadIdx.x >> 5;
  const int mtiles = (Nn + 15) >> 4;
  const int mg = blockIdx.x >> 3;       // group of 4 m-tiles
  const int ns = blockIdx.x & 7;        // 64-col super-tile

  // ---- stage packed B super-tile (16KB) into LDS with the Tensor Data Mover
  if (wave == 0) {
    unsigned long long ga = (unsigned long long)(const void*)(Wp + (size_t)ns * 8192);
    unsigned int lds = (unsigned int)(size_t)(void*)Bsh;   // low 32 bits = LDS addr
    v4u g0;
    g0[0] = 1u;                                        // count=1, user mode
    g0[1] = lds;                                       // lds_addr
    g0[2] = (unsigned int)(ga & 0xffffffffu);          // global_addr[31:0]
    g0[3] = (unsigned int)((ga >> 32) & 0x01ffffffu)   // global_addr[56:32]
            | (2u << 30);                              // type=2 (image)
    v8i g1;
    g1[0] = 0x20000;               // workgroup_mask=0, data_size=2 (4B units)
    g1[1] = (int)(4096u << 16);    // tensor_dim0[15:0]=4096 in bits[31:16]
    g1[2] = (int)(1u << 16);       // tensor_dim0 hi=0; tensor_dim1=1
    g1[3] = (int)(4096u << 16);    // tensor_dim1 hi=0; tile_dim0=4096
    g1[4] = 1;                     // tile_dim1=1, tile_dim2=0
    g1[5] = 4096;                  // tensor_dim0_stride[31:0]
    g1[6] = 0;                     // stride0 hi / stride1 lo
    g1[7] = 0;                     // stride1 hi
    v4i g2 = {0, 0, 0, 0};
    v4i g3 = {0, 0, 0, 0};
    v8i g4 = {0, 0, 0, 0, 0, 0, 0, 0};
    __builtin_amdgcn_tensor_load_to_lds(g0, g1, g2, g3, g4, 0);
    __builtin_amdgcn_s_wait_tensorcnt((short)0);
  }
  __syncthreads();

  const int mt = mg * 4 + wave;
  int mrow = mt * 16 + (lane & 15);
  if (mrow >= Nn) mrow = Nn - 1;          // clamp: loads always valid
  const int kb = (lane < 16) ? 0 : 16;    // A fragment k-base

  const v8f vzero = {0.f,0.f,0.f,0.f,0.f,0.f,0.f,0.f};
  v8f acc[4];
#pragma unroll
  for (int t = 0; t < 4; ++t) acc[t] = vzero;

  const __bf16* ap = xb + (size_t)mrow * DIN;

#pragma unroll
  for (int kblk = 0; kblk < 4; ++kblk) {
    const int kk = kblk * 32;
    v8bf alo = *(const v8bf*)(ap + kk + kb);
    v8bf ahi = *(const v8bf*)(ap + kk + kb + 8);
    v16bf a = __builtin_shufflevector(alo, ahi,
                                      0,1,2,3,4,5,6,7,8,9,10,11,12,13,14,15);
#pragma unroll
    for (int t = 0; t < 4; ++t) {
      v16bf b = *(const v16bf*)(Bsh + ((t * 4 + kblk) * 32 + lane) * 16);
      acc[t] = __builtin_amdgcn_wmma_f32_16x16x32_bf16(
          false, a, false, b, (short)0, acc[t], false, false);
    }
  }

#pragma unroll
  for (int t = 0; t < 4; ++t) {
    const int n = ns * 64 + t * 16 + (lane & 15);
#pragma unroll
    for (int r = 0; r < 8; ++r) {
      const int m = mt * 16 + r + ((lane < 16) ? 0 : 8);
      if (m < Nn) h[(size_t)m * HOUT + n] = acc[t][r];
    }
  }
}

// ---------------------------------------------------------------------------
// a_s[n,h] = <h[n,h,:], att_src[h,:]>, a_d likewise.  One wave per (n,head),
// float4 per lane (32 lanes x 4 = 128), shfl_xor tree reduce.
// ---------------------------------------------------------------------------
__global__ __launch_bounds__(256) void gat_attn(const float* __restrict__ hf,
                                                const float* __restrict__ att_src,
                                                const float* __restrict__ att_dst,
                                                float* __restrict__ a_s,
                                                float* __restrict__ a_d,
                                                int Nn) {
  const int lane = threadIdx.x & 31;
  const int gw = (blockIdx.x * blockDim.x + threadIdx.x) >> 5;
  if (gw >= Nn * NHEAD) return;
  const int n = gw >> 2, hh = gw & 3;
  const float4* hr4 = (const float4*)(hf + (size_t)n * HOUT + hh * DOUTC);
  const float4* as4 = (const float4*)(att_src + hh * DOUTC);
  const float4* ad4 = (const float4*)(att_dst + hh * DOUTC);
  float4 v = hr4[lane];
  float4 s = as4[lane];
  float4 d = ad4[lane];
  float ss = v.x*s.x + v.y*s.y + v.z*s.z + v.w*s.w;
  float sd = v.x*d.x + v.y*d.y + v.z*d.z + v.w*d.w;
#pragma unroll
  for (int off = 16; off > 0; off >>= 1) {
    ss += __shfl_xor(ss, off, 32);
    sd += __shfl_xor(sd, off, 32);
  }
  if (lane == 0) { a_s[gw] = ss; a_d[gw] = sd; }
}

// sign-split float atomic max (maps to native i32-max / u32-min atomics)
__device__ inline void atomicMaxF(float* addr, float val) {
  if (val >= 0.f)
    atomicMax((int*)addr, __float_as_int(val));
  else
    atomicMin((unsigned int*)addr, __float_as_uint(val));
}

// ---------------------------------------------------------------------------
// pass 1: per-edge leaky-relu logit; segment max over dst (edges + self-loops)
// ---------------------------------------------------------------------------
__global__ __launch_bounds__(256) void gat_edge_max(const int* __restrict__ ei,
                                                    const float* __restrict__ a_s,
                                                    const float* __restrict__ a_d,
                                                    float* __restrict__ mx,
                                                    float* __restrict__ elog,
                                                    int E, int Nn) {
  const int e = blockIdx.x * blockDim.x + threadIdx.x;
  if (e >= E + Nn) return;
  const int src = (e < E) ? ei[e]     : (e - E);
  const int dst = (e < E) ? ei[E + e] : (e - E);
  const float4 s4 = ((const float4*)a_s)[src];
  const float4 d4 = ((const float4*)a_d)[dst];
  float l[NHEAD] = { s4.x + d4.x, s4.y + d4.y, s4.z + d4.z, s4.w + d4.w };
#pragma unroll
  for (int hh = 0; hh < NHEAD; ++hh) {
    l[hh] = (l[hh] > 0.f) ? l[hh] : SLOPE * l[hh];
    atomicMaxF(&mx[dst * NHEAD + hh], l[hh]);
  }
  ((float4*)elog)[e] = make_float4(l[0], l[1], l[2], l[3]);
}

// ---------------------------------------------------------------------------
// pass 2: e = exp(logit - m[dst]); segment sum -> denom; elog <- e
// ---------------------------------------------------------------------------
__global__ __launch_bounds__(256) void gat_edge_sum(const int* __restrict__ ei,
                                                    const float* __restrict__ mx,
                                                    float* __restrict__ elog,
                                                    float* __restrict__ denom,
                                                    int E, int Nn) {
  const int e = blockIdx.x * blockDim.x + threadIdx.x;
  if (e >= E + Nn) return;
  const int dst = (e < E) ? ei[E + e] : (e - E);
  const float4 lg = ((const float4*)elog)[e];
  const float4 m4 = ((const float4*)mx)[dst];
  float w[NHEAD] = { __expf(lg.x - m4.x), __expf(lg.y - m4.y),
                     __expf(lg.z - m4.z), __expf(lg.w - m4.w) };
#pragma unroll
  for (int hh = 0; hh < NHEAD; ++hh)
    atomicAdd(&denom[dst * NHEAD + hh], w[hh]);
  ((float4*)elog)[e] = make_float4(w[0], w[1], w[2], w[3]);
}

// ---------------------------------------------------------------------------
// pass 3: out[dst,c] += sum_h (alpha_h/H) * h[src,h,c].  One wave per edge;
// lane covers 4 consecutive c (float4 loads, fully coalesced 16B strips).
// h (102.4 MB) is L2-resident on MI455X (192 MB L2) -> L2-bandwidth scatter.
// ---------------------------------------------------------------------------
__global__ __launch_bounds__(256) void gat_scatter(const int* __restrict__ ei,
                                                   const float* __restrict__ hf,
                                                   const float* __restrict__ elog,
                                                   const float* __restrict__ denom,
                                                   float* __restrict__ out,
                                                   int E, int Nn) {
  const int lane = threadIdx.x & 31;
  const int e = (int)(((long long)blockIdx.x * blockDim.x + threadIdx.x) >> 5);
  if (e >= E + Nn) return;
  const int src = (e < E) ? ei[e]     : (e - E);
  const int dst = (e < E) ? ei[E + e] : (e - E);

  const float4 w4 = ((const float4*)elog)[e];
  const float4 dn = ((const float4*)denom)[dst];
  const float a0 = 0.25f * w4.x / (dn.x + 1e-16f);
  const float a1 = 0.25f * w4.y / (dn.y + 1e-16f);
  const float a2 = 0.25f * w4.z / (dn.z + 1e-16f);
  const float a3 = 0.25f * w4.w / (dn.w + 1e-16f);

  const float4* h4 = (const float4*)(hf + (size_t)src * HOUT);
  const float4 v0 = h4[lane];
  const float4 v1 = h4[32 + lane];
  const float4 v2 = h4[64 + lane];
  const float4 v3 = h4[96 + lane];

  float* orow = out + (size_t)dst * DOUTC + lane * 4;
  atomicAdd(orow + 0, a0*v0.x + a1*v1.x + a2*v2.x + a3*v3.x);
  atomicAdd(orow + 1, a0*v0.y + a1*v1.y + a2*v2.y + a3*v3.y);
  atomicAdd(orow + 2, a0*v0.z + a1*v1.z + a2*v2.z + a3*v3.z);
  atomicAdd(orow + 3, a0*v0.w + a1*v1.w + a2*v2.w + a3*v3.w);
}

// ---------------------------------------------------------------------------
extern "C" void kernel_launch(void* const* d_in, const int* in_sizes, int n_in,
                              void* d_out, int out_size, void* d_ws, size_t ws_size,
                              hipStream_t stream) {
  const float* x       = (const float*)d_in[0];
  const int*   ei      = (const int*)  d_in[1];
  const float* W       = (const float*)d_in[2];
  const float* att_src = (const float*)d_in[3];
  const float* att_dst = (const float*)d_in[4];
  float* out = (float*)d_out;

  const int Nn = in_sizes[0] / DIN;   // 50000
  const int E  = in_sizes[1] / 2;     // 800000
  const int Et = E + Nn;

  // workspace layout
  float* ws     = (float*)d_ws;
  float* h_feat = ws;                              // Nn*512 f32
  float* a_s    = h_feat + (size_t)Nn * HOUT;      // Nn*4
  float* a_d    = a_s    + (size_t)Nn * NHEAD;     // Nn*4
  float* mx     = a_d    + (size_t)Nn * NHEAD;     // Nn*4
  float* denom  = mx     + (size_t)Nn * NHEAD;     // Nn*4
  float* elog   = denom  + (size_t)Nn * NHEAD;     // Et*4
  __bf16* xb    = (__bf16*)(elog + (size_t)Et * NHEAD);   // Nn*128 bf16
  __bf16* Wp    = xb + (size_t)Nn * DIN;                  // 128*512 bf16

  const int nOut = Nn * DOUTC;
  const int nNH  = Nn * NHEAD;
  gat_init<<<(nOut + 255) / 256, 256, 0, stream>>>(out, mx, denom, nOut, nNH);

  const int nx = Nn * DIN;
  gat_pack_x<<<(nx + 255) / 256, 256, 0, stream>>>(x, xb, nx);
  gat_pack_w<<<(DIN * HOUT + 255) / 256, 256, 0, stream>>>(W, Wp);

  const int mtiles = (Nn + 15) / 16;
  const int mgroups = (mtiles + 3) / 4;
  gat_gemm_wmma<<<mgroups * 8, 128, 0, stream>>>(xb, Wp, h_feat, Nn);

  const int pairs = Nn * NHEAD;
  gat_attn<<<(pairs + 7) / 8, 256, 0, stream>>>(h_feat, att_src, att_dst,
                                                a_s, a_d, Nn);

  gat_edge_max<<<(Et + 255) / 256, 256, 0, stream>>>(ei, a_s, a_d, mx, elog, E, Nn);
  gat_edge_sum<<<(Et + 255) / 256, 256, 0, stream>>>(ei, mx, elog, denom, E, Nn);
  gat_scatter<<<(Et + 7) / 8, 256, 0, stream>>>(ei, h_feat, elog, denom, out, E, Nn);
}
